// TripletLoss_81114752352813
// MI455X (gfx1250) — matrix operand order repaired and checked by reading, not compiled, not run on hardware
//
#include <hip/hip_runtime.h>
#include <float.h>

// ---------------------------------------------------------------------------
// TripletLoss on MI455X (gfx1250, wave32).
// Per (b, kind): masked mean over tokens of max over regions of T @ I^T.
// - fp32 -> bf16 hi/lo split; 3x V_WMMA_F32_16X16X32_BF16 per K=32 chunk
//   (a_hi*b_hi + a_hi*b_lo + a_lo*b_hi, fp32 accum) ~ fp32 accuracy at 8x the
//   K-depth per matrix issue of the fp32 16x16x4 WMMA.
// - masks are prefix masks (reference: arange < n); count them once and skip
//   invalid row/column tiles -> ~2.5x fewer WMMA issues on average.
// - two accumulators per inner iteration so the WMMA->VALU hazard slots of
//   tile i are filled by tile i+1's WMMA chain instead of v_nop.
// ---------------------------------------------------------------------------

typedef __attribute__((ext_vector_type(16))) __bf16 v16bf;
typedef __attribute__((ext_vector_type(8)))  __bf16 v8bf;
typedef __attribute__((ext_vector_type(8)))  float  v8f;
typedef __attribute__((ext_vector_type(4)))  float  v4f;

constexpr int kB = 128;   // batch
constexpr int kN = 1024;  // tokens
constexpr int kL = 512;   // regions
constexpr int kD = 64;    // feature dim
constexpr int kLSup   = 128;        // image columns staged per LDS supertile
constexpr int kLdsStr = 72;         // padded LDS stride (bf16 elems): 144B, bank-spread

__device__ __forceinline__ void splitf(float x, __bf16& hi, __bf16& lo) {
    // truncation split: hi exact (low 16 mantissa bits zeroed), lo = rounded residual
    unsigned u = __float_as_uint(x) & 0xffff0000u;
    float fh = __uint_as_float(u);
    hi = __builtin_bit_cast(__bf16, (unsigned short)(u >> 16));
    lo = (__bf16)(x - fh);
}

__global__ __launch_bounds__(256) void sim_kernel(
    const float* __restrict__ text, const float* __restrict__ img,
    const float* __restrict__ tmask, const float* __restrict__ imask,
    const int* __restrict__ iimp, const int* __restrict__ simp,
    float* __restrict__ sims)
{
    __shared__ __bf16 sHi[kLSup * kLdsStr];   // 18 KB image tile, bf16 hi
    __shared__ __bf16 sLo[kLSup * kLdsStr];   // 18 KB image tile, bf16 lo
    __shared__ float  sMax[kN];               // per-token running max over regions
    __shared__ float  sRed[16];
    __shared__ float  sDen;
    __shared__ int    sNtok, sNreg;

    const int b    = blockIdx.x;
    const int kind = blockIdx.y;              // 0=anchor, 1=I-impostor, 2=S-impostor
    int tb = b, ib = b;
    if (kind == 1)      ib = iimp[b];
    else if (kind == 2) tb = simp[b];

    const float* __restrict__ Tp = text  + (size_t)tb * kN * kD;
    const float* __restrict__ Ip = img   + (size_t)ib * kL * kD;
    const float* __restrict__ tm = tmask + (size_t)tb * kN;
    const float* __restrict__ im = imask + (size_t)ib * kL;

    const int tid  = threadIdx.x;
    const int lane = tid & 31;
    const int wave = tid >> 5;
    const int hf   = (lane >> 4) & 1;         // which 16-lane half of the wave
    const int l15  = lane & 15;

    // ---- count valid tokens / regions (prefix masks) and init row maxes ----
    float tcnt = 0.f, icnt = 0.f;
    for (int r = tid; r < kN; r += 256) { tcnt += tm[r]; sMax[r] = -FLT_MAX; }
    for (int r = tid; r < kL; r += 256) icnt += im[r];
    #pragma unroll
    for (int off = 16; off; off >>= 1) {
        tcnt += __shfl_xor(tcnt, off, 32);
        icnt += __shfl_xor(icnt, off, 32);
    }
    if (lane == 0) { sRed[wave] = tcnt; sRed[8 + wave] = icnt; }
    __syncthreads();
    if (tid == 0) {
        float ts = 0.f, is = 0.f;
        #pragma unroll
        for (int w = 0; w < 8; ++w) { ts += sRed[w]; is += sRed[8 + w]; }
        sDen  = ts;
        sNtok = (int)(ts + 0.5f);
        sNreg = (int)(is + 0.5f);
    }
    __syncthreads();
    const int ntok = sNtok;
    const int nreg = sNreg;
    const int nSup = (nreg + kLSup - 1) / kLSup;

    for (int st = 0; st < nSup; ++st) {
        const int l0 = st * kLSup;

        // ---- stage 128 image columns into LDS as bf16 hi/lo -------------
        {
            const int c  = tid >> 1;
            const int d0 = (tid & 1) * 32;
            const float* src = Ip + (size_t)(l0 + c) * kD + d0;
            __bf16* dh = sHi + c * kLdsStr + d0;
            __bf16* dl = sLo + c * kLdsStr + d0;
            #pragma unroll
            for (int j = 0; j < 32; j += 4) {
                v4f x = *(const v4f*)(src + j);
                #pragma unroll
                for (int k = 0; k < 4; ++k) {
                    __bf16 h, l;
                    splitf(x[k], h, l);
                    dh[j + k] = h;
                    dl[j + k] = l;
                }
            }
            if (st + 1 < nSup)   // pull next supertile toward L2/L0
                __builtin_prefetch(Ip + (size_t)(l0 + kLSup + c) * kD + d0, 0, 0);
        }
        __syncthreads();

        // number of 16-col tiles actually containing valid regions
        const int nct = min(kLSup / 16, (nreg - l0 + 15) >> 4);

        // ---- each wave: its 16-row tiles, skipping tiles past ntok -------
        for (int rt = 0;; ++rt) {
            const int m0 = (rt * 8 + wave) * 16;
            if (m0 >= ntok) break;
            const int row = m0 + l15;

            // A operand in ISA 16-bit A layout: lane half 0 -> K {0..7,16..23},
            // half 1 -> K {8..15,24..31} (per K=32 chunk)
            v16bf aHi[2], aLo[2];
            #pragma unroll
            for (int kc = 0; kc < 2; ++kc) {
                const float* ap = Tp + (size_t)row * kD + kc * 32 + hf * 8;
                #pragma unroll
                for (int jj = 0; jj < 8; jj += 4) {
                    v4f x0 = *(const v4f*)(ap + jj);
                    v4f x1 = *(const v4f*)(ap + 16 + jj);
                    #pragma unroll
                    for (int k = 0; k < 4; ++k) {
                        __bf16 h, l;
                        splitf(x0[k], h, l); aHi[kc][jj + k]     = h; aLo[kc][jj + k]     = l;
                        splitf(x1[k], h, l); aHi[kc][8 + jj + k] = h; aLo[kc][8 + jj + k] = l;
                    }
                }
            }

            v8f mx;
            #pragma unroll
            for (int r = 0; r < 8; ++r) mx[r] = -FLT_MAX;

            // one 16x16 tile: B from LDS (per lane K-contiguous), 6 WMMAs
            auto mm_tile = [&](int ctIdx, v8f& acc) {
                const int cc = ctIdx * 16 + l15;      // this lane's column
                #pragma unroll
                for (int kc = 0; kc < 2; ++kc) {
                    const __bf16* bh = sHi + cc * kLdsStr + kc * 32 + hf * 8;
                    const __bf16* bl = sLo + cc * kLdsStr + kc * 32 + hf * 8;
                    v8bf t0 = *(const v8bf*)(bh);
                    v8bf t1 = *(const v8bf*)(bh + 16);
                    v8bf u0 = *(const v8bf*)(bl);
                    v8bf u1 = *(const v8bf*)(bl + 16);
                    v16bf bHi, bLo;
                    #pragma unroll
                    for (int k = 0; k < 8; ++k) {
                        bHi[k] = t0[k]; bHi[8 + k] = t1[k];
                        bLo[k] = u0[k]; bLo[8 + k] = u1[k];
                    }
                    acc = __builtin_amdgcn_wmma_f32_16x16x32_bf16(
                        false, aLo[kc], false, bHi, (short)0, acc, false, false);
                    acc = __builtin_amdgcn_wmma_f32_16x16x32_bf16(
                        false, aHi[kc], false, bLo, (short)0, acc, false, false);
                    acc = __builtin_amdgcn_wmma_f32_16x16x32_bf16(
                        false, aHi[kc], false, bHi, (short)0, acc, false, false);
                }
            };
            auto max_tile = [&](int ctIdx, const v8f& acc) {
                const bool valid = (l0 + ctIdx * 16 + l15) < nreg;  // prefix mask
                #pragma unroll
                for (int r = 0; r < 8; ++r) {
                    float v = valid ? acc[r] : -FLT_MAX;
                    mx[r] = fmaxf(mx[r], v);
                }
            };

            int ct = 0;
            for (; ct + 2 <= nct; ct += 2) {
                // two independent accumulator chains: tile ct+1's WMMAs fill
                // tile ct's WMMA->VALU hazard slots
                v8f acc0 = {}, acc1 = {};
                mm_tile(ct,     acc0);
                mm_tile(ct + 1, acc1);
                max_tile(ct,     acc0);
                max_tile(ct + 1, acc1);
            }
            if (ct < nct) {
                v8f acc0 = {};
                mm_tile(ct, acc0);
                max_tile(ct, acc0);
            }

            // reduce max across the 16 lanes of each half (C layout: reg r = row r / r+8)
            #pragma unroll
            for (int off = 1; off < 16; off <<= 1) {
                #pragma unroll
                for (int r = 0; r < 8; ++r)
                    mx[r] = fmaxf(mx[r], __shfl_xor(mx[r], off, 32));
            }
            if (l15 == 0) {
                const int base = m0 + hf * 8;   // rows exclusive to this wave: no race
                #pragma unroll
                for (int r = 0; r < 8; ++r)
                    sMax[base + r] = fmaxf(sMax[base + r], mx[r]);
            }
        }
        __syncthreads();
    }

    // ---- masked mean over tokens -> sim -------------------------------
    float num = 0.f;
    for (int r = tid; r < ntok; r += 256)
        num += sMax[r] * tm[r];
    #pragma unroll
    for (int off = 16; off; off >>= 1)
        num += __shfl_xor(num, off, 32);
    if (lane == 0) sRed[wave] = num;
    __syncthreads();
    if (tid == 0) {
        float n = 0.f;
        #pragma unroll
        for (int w = 0; w < 8; ++w) n += sRed[w];
        sims[b * 3 + kind] = n / sDen;
    }
}

__global__ __launch_bounds__(128) void loss_kernel(const float* __restrict__ sims,
                                                   float* __restrict__ out)
{
    __shared__ float sRed[4];
    const int b = threadIdx.x;
    const float anchor = sims[b * 3 + 0];
    const float ii     = sims[b * 3 + 1];
    const float si     = sims[b * 3 + 2];
    float per = fmaxf(0.f, 1.f + ii - anchor) + fmaxf(0.f, 1.f + si - anchor);
    #pragma unroll
    for (int off = 16; off; off >>= 1) per += __shfl_xor(per, off, 32);
    if ((b & 31) == 0) sRed[b >> 5] = per;
    __syncthreads();
    if (b == 0) {
        float s = 0.f;
        #pragma unroll
        for (int w = 0; w < 4; ++w) s += sRed[w];
        out[0] = s / (float)kB;
    }
}

extern "C" void kernel_launch(void* const* d_in, const int* in_sizes, int n_in,
                              void* d_out, int out_size, void* d_ws, size_t ws_size,
                              hipStream_t stream) {
    (void)in_sizes; (void)n_in; (void)out_size; (void)ws_size;
    const float* text = (const float*)d_in[0];
    const float* img  = (const float*)d_in[1];
    const float* tm   = (const float*)d_in[2];
    const float* im   = (const float*)d_in[3];
    const int*   iimp = (const int*)d_in[4];
    const int*   simp = (const int*)d_in[5];
    float* sims = (float*)d_ws;             // 3*B floats of scratch

    dim3 grid(kB, 3);
    sim_kernel<<<grid, 256, 0, stream>>>(text, img, tm, im, iimp, simp, sims);
    loss_kernel<<<1, 128, 0, stream>>>(sims, (float*)d_out);
}
